// CrossAttention_25847113187983
// MI455X (gfx1250) — matrix operand order repaired and checked by reading, not compiled, not run on hardware
//
#include <hip/hip_runtime.h>

// ---------------------------------------------------------------------------
// CrossAttention on MI455X (gfx1250, wave32, WMMA f32_16x16x32_f16 + TDM)
//
// B=4, C=256, N=H*W=4096.
//   S      = (W @ a)^T @ b          per batch: [N,N], K=C
//   a_new  = a @ softmax(S, axis=n) per batch: [C,N], K=N
//   b_new  = b @ softmax(S, axis=m)^T         [C,N], K=N
//
// Materialize S and S^T in f16 (2 x 134 MB; per-batch 33.5 MB slices are
// L2-resident during the big GEMMs), LDS-reduce softmax stats, transform S
// in place to probability matrices, then two pure-f16 WMMA GEMMs.
//
// GEMMs: 8-wave workgroups, 128x128 output tile, K staged in 128x32 LDS
// panels (80 B padded pitch -> conflict-free ds_load_b128 fragments),
// double buffered.  Panel staging uses the Tensor Data Mover
// (tensor_load_to_lds, TENSORcnt) issued by wave 0 while all waves run
// WMMA; falls back to manual global->LDS copies if the builtin is absent.
// Workspace needed: ~289 MB.
// ---------------------------------------------------------------------------

typedef _Float16 v16h __attribute__((ext_vector_type(16)));
typedef _Float16 v8h  __attribute__((ext_vector_type(8)));
typedef float    v8f  __attribute__((ext_vector_type(8)));

static constexpr int B_ = 4;
static constexpr int C_ = 256;
static constexpr int N_ = 4096;   // 64*64

// LDS panel geometry: 128 rows x 32 halfs, padded pitch 40 halfs (80 B)
static constexpr int PITCH   = 40;      // halfs
static constexpr int PANELH  = 128 * PITCH;          // 5120 halfs / panel
static constexpr int BOFF    = 2 * PANELH;           // B panels after A's
static constexpr int SMEMSZ  = 4 * PANELH;           // 20480 halfs = 40 KB

#if defined(__HIP_DEVICE_COMPILE__) && defined(__gfx1250__) && \
    __has_builtin(__builtin_amdgcn_tensor_load_to_lds)
#define USE_TDM 1
#else
#define USE_TDM 0
#endif

__device__ __forceinline__ v8f wmma_f16(v16h a, v16h b, v8f c) {
    return __builtin_amdgcn_wmma_f32_16x16x32_f16(
        false, a, false, b, (short)0, c, false, false);
}

#if USE_TDM
// Issue one TDM 2-D tile load: 128 rows x 32 halfs from a row-major f16
// matrix with leading dimension K (halfs), into LDS at lds_byte_addr with
// +4-DWORD padding after every 16 DWORDs (row pitch 40 halfs).
// D# layout per CDNA5 ISA ch.8 (group0: count/lds/global/type; group1:
// data_size, pad, tensor_dim0=K, tensor_dim1=big, tile 32x128, stride=K).
__device__ __forceinline__ void tdm_load_panel(unsigned lds_byte_addr,
                                               const _Float16* gsrc, int K) {
    typedef unsigned uint32x4 __attribute__((ext_vector_type(4)));
    typedef int      int32x4  __attribute__((ext_vector_type(4)));
    typedef int      int32x8  __attribute__((ext_vector_type(8)));
    unsigned long long ga = (unsigned long long)(size_t)gsrc;
    uint32x4 g0;
    g0.x = 1u;                                  // count=1, user descriptor
    g0.y = lds_byte_addr;                       // lds_addr
    g0.z = (unsigned)ga;                        // global_addr[31:0]
    g0.w = (unsigned)((ga >> 32) & 0x01FFFFFFull) | (2u << 30);  // type=2
    int32x8 g1;
    // data_size=1(2B)<<16 | pad_enable<<20 | pad_interval=3(16 dw)<<22
    // | pad_amount=3(4 dw)<<25
    g1[0] = 0x06D10000;
    g1[1] = (K & 0xFFFF) << 16;                 // tensor_dim0[15:0]
    g1[2] = (K >> 16) & 0xFFFF;                 // dim0[31:16] | dim1[15:0]=0
    g1[3] = 16 | (32 << 16);                    // dim1=1<<20, tile_dim0=32
    g1[4] = 128;                                // tile_dim1=128, tile_dim2=0
    g1[5] = K;                                  // tensor_dim0_stride[31:0]
    g1[6] = 0;
    g1[7] = 0;
    int32x4 z4 = {0, 0, 0, 0};
#if __clang_major__ >= 23
    int32x8 z8 = {0, 0, 0, 0, 0, 0, 0, 0};
    __builtin_amdgcn_tensor_load_to_lds(g0, g1, z4, z4, z8, 0);
#else
    __builtin_amdgcn_tensor_load_to_lds(g0, g1, z4, z4, 0);
#endif
}
#endif

// ---- 1. f32 -> f16 conversions (a16, b16, bT16) ---------------------------
__global__ void convert_kernel(const float* __restrict__ a,
                               const float* __restrict__ b,
                               _Float16* __restrict__ a16,
                               _Float16* __restrict__ b16,
                               _Float16* __restrict__ bT16) {
    size_t i = (size_t)blockIdx.x * blockDim.x + threadIdx.x;  // < B*C*N
    float av = a[i], bv = b[i];
    a16[i] = (_Float16)av;
    b16[i] = (_Float16)bv;
    int m = (int)(i % N_);
    size_t t = i / N_;
    int d = (int)(t % C_);
    int bb = (int)(t / C_);
    bT16[((size_t)bb * N_ + m) * C_ + d] = (_Float16)bv;
}

// ---- 2. WaT16[bb][n][d] = sum_c W[d][c] * a[bb][c][n] ---------------------
__global__ void waT_kernel(const float* __restrict__ a,
                           const float* __restrict__ W,
                           _Float16* __restrict__ WaT) {
    int n  = blockIdx.x * blockDim.x + threadIdx.x;
    int d  = blockIdx.y;
    int bb = blockIdx.z;
    const float* acol = a + (size_t)bb * C_ * N_ + n;   // stride N_ over c
    const float* wrow = W + (size_t)d * C_;
    float acc = 0.f;
#pragma unroll 8
    for (int c = 0; c < C_; ++c)
        acc = fmaf(wrow[c], acol[(size_t)c * N_], acc);
    WaT[((size_t)bb * N_ + n) * C_ + d] = (_Float16)acc;
}

// ---- 3. Tiled WMMA GEMM: out = A[rows,K] @ B, with BT[cols,K] given -------
// A and BT both row-major with leading dimension == K.
// SMODE: write f16 result to S (row-major) and ST (transposed, via LDS).
// else : write f32 result to outF (rows = C_, ld = N_).
template<bool SMODE>
__global__ __launch_bounds__(256)
void tile_gemm_kernel(const _Float16* __restrict__ A,
                      const _Float16* __restrict__ BT,
                      const int K,
                      const size_t aStride, const size_t bStride,
                      _Float16* __restrict__ S, _Float16* __restrict__ ST,
                      float* __restrict__ outF) {
    __shared__ __attribute__((aligned(16))) _Float16 smem[SMEMSZ];  // 40 KB

    const int tid  = threadIdx.x;
    const int lane = tid & 31, l15 = lane & 15, hi = (lane >> 4) & 1;
    const int w    = tid >> 5;       // wave 0..7
    const int wr   = w & 1;          // row half of tile (64 rows)
    const int wc   = w >> 1;         // col quarter of tile (32 cols)
    const int bb   = blockIdx.z;

    const size_t rowBase = (size_t)blockIdx.x * 128;
    const size_t colBase = (size_t)blockIdx.y * 128;
    const _Float16* Ab = A  + (size_t)bb * aStride + rowBase * (size_t)K;
    const _Float16* Bb = BT + (size_t)bb * bStride + colBase * (size_t)K;

    v8f acc[4][2] = {};

    // one k-step of WMMA from LDS buffer p (conflict-free 80 B pitch)
    auto compute_step = [&](int p) {
        const _Float16* lA = smem + p * PANELH;
        const _Float16* lB = smem + BOFF + p * PANELH;
        v16h bfr[2];
#pragma unroll
        for (int ct = 0; ct < 2; ++ct) {
            const _Float16* bp =
                lB + (wc * 32 + ct * 16 + l15) * PITCH + hi * 16;
            union { v16h v; v8h h[2]; } f;
            f.h[0] = *(const v8h*)(bp);
            f.h[1] = *(const v8h*)(bp + 8);
            bfr[ct] = f.v;
        }
#pragma unroll
        for (int rt = 0; rt < 4; ++rt) {
            const _Float16* ap =
                lA + (wr * 64 + rt * 16 + l15) * PITCH + hi * 16;
            union { v16h v; v8h h[2]; } f;
            f.h[0] = *(const v8h*)(ap);
            f.h[1] = *(const v8h*)(ap + 8);
            acc[rt][0] = wmma_f16(f.v, bfr[0], acc[rt][0]);
            acc[rt][1] = wmma_f16(f.v, bfr[1], acc[rt][1]);
        }
    };

    const int steps = K >> 5;
    int p = 0;

#if USE_TDM
    // --- TDM-staged double buffering: wave 0 drives the DMA engine --------
    const unsigned ldsBase = (unsigned)(size_t)(&smem[0]);
    if (w == 0) {
        tdm_load_panel(ldsBase,                 Ab, K);
        tdm_load_panel(ldsBase + 2 * BOFF,      Bb, K);   // byte offset
        __builtin_amdgcn_s_wait_tensorcnt(0);
    }
    __syncthreads();
    for (int kk = 0; kk < steps; ++kk) {
        const bool hasNext = (kk + 1) < steps;
        if (hasNext && w == 0) {   // DMA next panels while everyone computes
            const int k0 = (kk + 1) << 5;
            tdm_load_panel(ldsBase + (1 - p) * 2 * PANELH, Ab + k0, K);
            tdm_load_panel(ldsBase + 2 * BOFF + (1 - p) * 2 * PANELH,
                           Bb + k0, K);
        }
        compute_step(p);
        if (hasNext && w == 0) __builtin_amdgcn_s_wait_tensorcnt(0);
        __syncthreads();
        p ^= 1;
    }
#else
    // --- fallback: cooperative copy with register prefetch ----------------
    const int cprow = tid >> 1;
    const int cpseg = (tid & 1) * 16;
    v8h ra0, ra1, rb0, rb1;
    {
        const _Float16* pA = Ab + (size_t)cprow * K + cpseg;
        const _Float16* pB = Bb + (size_t)cprow * K + cpseg;
        ra0 = *(const v8h*)pA; ra1 = *(const v8h*)(pA + 8);
        rb0 = *(const v8h*)pB; rb1 = *(const v8h*)(pB + 8);
        _Float16* dA = smem +        cprow * PITCH + cpseg;
        _Float16* dB = smem + BOFF + cprow * PITCH + cpseg;
        *(v8h*)dA = ra0; *(v8h*)(dA + 8) = ra1;
        *(v8h*)dB = rb0; *(v8h*)(dB + 8) = rb1;
    }
    __syncthreads();
    for (int kk = 0; kk < steps; ++kk) {
        const bool hasNext = (kk + 1) < steps;
        if (hasNext) {
            const int k0 = (kk + 1) << 5;
            const _Float16* pA = Ab + (size_t)cprow * K + k0 + cpseg;
            const _Float16* pB = Bb + (size_t)cprow * K + k0 + cpseg;
            ra0 = *(const v8h*)pA; ra1 = *(const v8h*)(pA + 8);
            rb0 = *(const v8h*)pB; rb1 = *(const v8h*)(pB + 8);
        }
        compute_step(p);
        __syncthreads();
        if (hasNext) {
            _Float16* dA = smem + (1 - p) * PANELH + cprow * PITCH + cpseg;
            _Float16* dB = smem + BOFF + (1 - p) * PANELH
                                + cprow * PITCH + cpseg;
            *(v8h*)dA = ra0; *(v8h*)(dA + 8) = ra1;
            *(v8h*)dB = rb0; *(v8h*)(dB + 8) = rb1;
        }
        __syncthreads();
        p ^= 1;
    }
#endif

    // C/D layout: VGPR j -> row j (lanes 0-15) / j+8 (lanes 16-31), col = l15
    if constexpr (SMODE) {
        // row-major S: direct f16 stores (16-lane x 2B contiguous runs)
#pragma unroll
        for (int rt = 0; rt < 4; ++rt)
#pragma unroll
            for (int ct = 0; ct < 2; ++ct)
#pragma unroll
                for (int j = 0; j < 8; ++j) {
                    const int r = wr * 64 + rt * 16 + j + hi * 8;
                    const int c = wc * 32 + ct * 16 + l15;
                    S[((size_t)bb * N_ + rowBase + r) * N_ + colBase + c] =
                        (_Float16)acc[rt][ct][j];
                }
        // transposed ST: stage 128x128 f16 tile in LDS, store coalesced
        __syncthreads();
#pragma unroll
        for (int rt = 0; rt < 4; ++rt)
#pragma unroll
            for (int ct = 0; ct < 2; ++ct)
#pragma unroll
                for (int j = 0; j < 8; ++j) {
                    const int r = wr * 64 + rt * 16 + j + hi * 8;
                    const int c = wc * 32 + ct * 16 + l15;
                    smem[c * 128 + r] = (_Float16)acc[rt][ct][j];
                }
        __syncthreads();
        const int trow = tid >> 1;             // transposed row = col of S
        const int tseg = (tid & 1) * 64;       // 64-half (128 B) segment
        const _Float16* src = smem + trow * 128 + tseg;
        _Float16* dst = ST + ((size_t)bb * N_ + colBase + trow) * N_
                           + rowBase + tseg;
#pragma unroll
        for (int q = 0; q < 8; ++q)
            ((v8h*)dst)[q] = ((const v8h*)src)[q];
    } else {
        float* ob = outF + (size_t)bb * C_ * N_;
#pragma unroll
        for (int rt = 0; rt < 4; ++rt)
#pragma unroll
            for (int ct = 0; ct < 2; ++ct)
#pragma unroll
                for (int j = 0; j < 8; ++j) {
                    const int r = wr * 64 + rt * 16 + j + hi * 8;
                    const int c = wc * 32 + ct * 16 + l15;
                    ob[(rowBase + r) * (size_t)N_ + colBase + c] =
                        acc[rt][ct][j];
                }
    }
}

// ---- 4. per-row max & sum(exp) over rows of a [B*N, N] f16 matrix ---------
__global__ void stats_kernel(const _Float16* __restrict__ buf,
                             float* __restrict__ mx,
                             float* __restrict__ sm) {
    __shared__ float red[256];
    const int row = blockIdx.x;           // B*N rows
    const int tid = threadIdx.x;          // 256 threads, 16 elems each
    const _Float16* p = buf + (size_t)row * N_ + tid * 16;
    union { v8h h[2]; _Float16 e[16]; } d;
    d.h[0] = *(const v8h*)(p);
    d.h[1] = *(const v8h*)(p + 8);
    float lmax = -3.4e38f;
#pragma unroll
    for (int j = 0; j < 16; ++j) lmax = fmaxf(lmax, (float)d.e[j]);
    red[tid] = lmax; __syncthreads();
    for (int s = 128; s > 0; s >>= 1) {
        if (tid < s) red[tid] = fmaxf(red[tid], red[tid + s]);
        __syncthreads();
    }
    const float m = red[0]; __syncthreads();
    float lsum = 0.f;
#pragma unroll
    for (int j = 0; j < 16; ++j) lsum += __expf((float)d.e[j] - m);
    red[tid] = lsum; __syncthreads();
    for (int s = 128; s > 0; s >>= 1) {
        if (tid < s) red[tid] += red[tid + s];
        __syncthreads();
    }
    if (tid == 0) { mx[row] = m; sm[row] = red[0]; }
}

// ---- 5. in-place buf[row][j] = exp(buf - mx[row]) / sm[row] ---------------
__global__ void transform_kernel(_Float16* __restrict__ buf,
                                 const float* __restrict__ mx,
                                 const float* __restrict__ sm) {
    const int row = blockIdx.x;
    const int tid = threadIdx.x;
    const float m = mx[row];
    const float r = 1.0f / sm[row];
    _Float16* p = buf + (size_t)row * N_ + tid * 16;
    union { v8h h[2]; _Float16 e[16]; } d;
    d.h[0] = *(const v8h*)(p);
    d.h[1] = *(const v8h*)(p + 8);
#pragma unroll
    for (int j = 0; j < 16; ++j)
        d.e[j] = (_Float16)(__expf((float)d.e[j] - m) * r);
    *(v8h*)(p)     = d.h[0];
    *(v8h*)(p + 8) = d.h[1];
}

// ---------------------------------------------------------------------------
extern "C" void kernel_launch(void* const* d_in, const int* in_sizes, int n_in,
                              void* d_out, int out_size, void* d_ws, size_t ws_size,
                              hipStream_t stream) {
    const float* a = (const float*)d_in[0];   // [B,C,H,W]
    const float* b = (const float*)d_in[1];   // [B,C,H,W]
    const float* W = (const float*)d_in[2];   // [C,C]
    float* out = (float*)d_out;               // a_new ++ b_new (f32)

    const size_t elems = (size_t)B_ * C_ * N_;        // 4,194,304
    const size_t snn   = (size_t)B_ * N_ * N_;        // 67,108,864
    char* ws = (char*)d_ws;

    _Float16* a16  = (_Float16*)(ws);
    _Float16* b16  = a16 + elems;
    _Float16* bT16 = b16 + elems;
    _Float16* WaT  = bT16 + elems;
    float* rowmax = (float*)(WaT + elems);
    float* rowsum = rowmax + (size_t)B_ * N_;
    float* colmax = rowsum + (size_t)B_ * N_;
    float* colsum = colmax + (size_t)B_ * N_;
    _Float16* S16  = (_Float16*)(colsum + (size_t)B_ * N_);
    _Float16* ST16 = S16 + snn;
    // total ws use: 4*elems*2 + 4*B*N*4 + 2*snn*2 ~= 289 MB

    // 1. conversions
    convert_kernel<<<dim3((unsigned)(elems / 256)), 256, 0, stream>>>(
        a, b, a16, b16, bT16);

    // 2. WaT = (W @ a)^T, stored [B][N][C] f16
    waT_kernel<<<dim3(N_ / 256, C_, B_), 256, 0, stream>>>(a, W, WaT);

    // 3. S = WaT @ b (tiled WMMA, K=C), write S and S^T in f16
    tile_gemm_kernel<true><<<dim3(N_ / 128, N_ / 128, B_), 256, 0, stream>>>(
        WaT, bT16, C_, (size_t)N_ * C_, (size_t)N_ * C_, S16, ST16, nullptr);

    // 4. softmax stats: rows of S -> axis-m stats; rows of S^T -> axis-n
    stats_kernel<<<dim3(B_ * N_), 256, 0, stream>>>(S16,  rowmax, rowsum);
    stats_kernel<<<dim3(B_ * N_), 256, 0, stream>>>(ST16, colmax, colsum);

    // 5. in-place transforms: S16 -> P2 (softmax over m), ST16 -> P1^T (over n)
    transform_kernel<<<dim3(B_ * N_), 256, 0, stream>>>(S16,  rowmax, rowsum);
    transform_kernel<<<dim3(B_ * N_), 256, 0, stream>>>(ST16, colmax, colsum);

    // 6. a_new = a16 @ P1  (B-operand column-major = rows of P1^T = ST16)
    tile_gemm_kernel<false><<<dim3(C_ / 128, N_ / 128, B_), 256, 0, stream>>>(
        a16, ST16, N_, (size_t)C_ * N_, (size_t)N_ * N_, nullptr, nullptr, out);
    //    b_new = b16 @ P2^T (B-operand column-major = rows of P2 = S16)
    tile_gemm_kernel<false><<<dim3(C_ / 128, N_ / 128, B_), 256, 0, stream>>>(
        b16, S16, N_, (size_t)C_ * N_, (size_t)N_ * N_, nullptr, nullptr,
        out + elems);
}
